// lstm_73237782331617
// MI455X (gfx1250) — compile-verified, matrix operand order checked by reading, not run on hardware
//
#include <hip/hip_runtime.h>

// CDNA5 / gfx1250, wave32.
typedef __attribute__((ext_vector_type(16))) _Float16 v16h;
typedef __attribute__((ext_vector_type(8)))  float    v8f;

#define T_LEN 4096
#define LOG2E   1.44269504088896340736f
#define LOG2E_2 2.88539008177792681472f   // 2*log2(e), folded into tanh

__device__ __forceinline__ float fast_exp(float x) {
  // v_exp_f32 computes 2^x; scale by log2(e)
  return __builtin_amdgcn_exp2f(x * LOG2E);
}
__device__ __forceinline__ float fast_sig(float x) {
  return __builtin_amdgcn_rcpf(1.0f + __builtin_amdgcn_exp2f(x * -LOG2E));
}
__device__ __forceinline__ float fast_tanh(float x) {
  float e = __builtin_amdgcn_exp2f(x * LOG2E_2);   // exp(2x)
  return (e - 1.0f) * __builtin_amdgcn_rcpf(e + 1.0f);
}

// Serial LSTM recurrence over T steps, H=2, gates precomputed in xp[t][0..7]
// (PyTorch order i,f,g,o). Runs on a single lane; software-pipelined so the
// ds_load of step t+1's gates overlaps step t's transcendental chain.
__device__ void lstm_recurrence(const float* __restrict__ xp,
                                float* __restrict__ hseq,
                                const float* __restrict__ Whh,  // [8][2]
                                float* hout0, float* hout1) {
  float wh[16];
#pragma unroll
  for (int j = 0; j < 16; ++j) wh[j] = Whh[j];
  float h0 = 0.f, h1 = 0.f, c0 = 0.f, c1 = 0.f;

  const float4* gp = (const float4*)xp;
  float4 ga = gp[0];  // gates 0..3 of step 0 (i0 i1 f0 f1)
  float4 gb = gp[1];  // gates 4..7 of step 0 (g0 g1 o0 o1)

  for (int t = 0; t < T_LEN; ++t) {
    // Prefetch next step's gate pre-activations (branch-free clamped index);
    // these do NOT depend on h/c, so the LDS latency hides under the math.
    const int tn = (t + 1) & (T_LEN - 1);
    float4 gna = gp[tn * 2 + 0];
    float4 gnb = gp[tn * 2 + 1];

    float q0 = fmaf(wh[1],  h1, fmaf(wh[0],  h0, ga.x));
    float q1 = fmaf(wh[3],  h1, fmaf(wh[2],  h0, ga.y));
    float q2 = fmaf(wh[5],  h1, fmaf(wh[4],  h0, ga.z));
    float q3 = fmaf(wh[7],  h1, fmaf(wh[6],  h0, ga.w));
    float q4 = fmaf(wh[9],  h1, fmaf(wh[8],  h0, gb.x));
    float q5 = fmaf(wh[11], h1, fmaf(wh[10], h0, gb.y));
    float q6 = fmaf(wh[13], h1, fmaf(wh[12], h0, gb.z));
    float q7 = fmaf(wh[15], h1, fmaf(wh[14], h0, gb.w));
    float i0 = fast_sig(q0),  i1 = fast_sig(q1);
    float f0 = fast_sig(q2),  f1 = fast_sig(q3);
    float g0 = fast_tanh(q4), g1 = fast_tanh(q5);
    float o0 = fast_sig(q6),  o1 = fast_sig(q7);
    c0 = fmaf(f0, c0, i0 * g0);
    c1 = fmaf(f1, c1, i1 * g1);
    h0 = o0 * fast_tanh(c0);
    h1 = o1 * fast_tanh(c1);
    hseq[t * 2 + 0] = h0;
    hseq[t * 2 + 1] = h1;

    ga = gna;
    gb = gnb;
  }
  *hout0 = h0;
  *hout1 = h1;
}

__global__ __launch_bounds__(256) void lstm5_kernel(
    const float* __restrict__ x,                                     // [4096,4096,1]
    const float* __restrict__ Wih0, const float* __restrict__ Whh0,  // [8,1], [8,2]
    const float* __restrict__ bih0, const float* __restrict__ bhh0,  // [8], [8]
    const float* __restrict__ Wih_r, const float* __restrict__ Whh_r,// [4,8,2] each
    const float* __restrict__ bih_r, const float* __restrict__ bhh_r,// [4,8] each
    float* __restrict__ out) {
  extern __shared__ float lds[];
  float* xp   = lds;              // [4096][8] gate pre-activations (128 KiB)
  float* hseq = lds + T_LEN * 8;  // [4096][2] hidden sequence       (32 KiB)

  const int tid  = threadIdx.x;
  const int lane = tid & 31;
  const int wave = tid >> 5;

  const float* xrow = x + (size_t)(T_LEN - 1) * T_LEN;  // only batch row 4095 matters

  // Warm the caches for the one row we touch (global_prefetch_b8): 256 threads
  // x 64B stride covers the whole 16 KiB row.
  __builtin_prefetch(xrow + tid * 16, 0, 1);

  // ---- Layer 0 input projection (input size 1): xp[t][g] = x[t]*Wih0[g] + b[g]
  {
    float w0[8], b0[8];
#pragma unroll
    for (int j = 0; j < 8; ++j) { w0[j] = Wih0[j]; b0[j] = bih0[j] + bhh0[j]; }
    for (int t = tid; t < T_LEN; t += blockDim.x) {
      float xv = xrow[t];
#pragma unroll
      for (int j = 0; j < 8; ++j) xp[t * 8 + j] = fmaf(xv, w0[j], b0[j]);
    }
  }
  __syncthreads();

  float h0 = 0.f, h1 = 0.f;
  if (tid == 0) lstm_recurrence(xp, hseq, Whh0, &h0, &h1);
  __syncthreads();

  // ---- Layers 1..4: WMMA input projection  xp = hseq @ Wih^T + b  , then recurrence
  for (int l = 0; l < 4; ++l) {
    const float* Wih = Wih_r + l * 16;  // [8][2]
    const float* bih = bih_r + l * 8;
    const float* bhh = bhh_r + l * 8;

    // B matrix (K x N = 32 x 16; only K<2, N<8 nonzero). Lane n holds column
    // N=n (lanes 0..15), elements = K 0..15; lanes 16..31 = K 16..31 (all zero).
    const int col  = lane & 15;
    const int wsel = lane & 7;            // clamped index -> always-valid load
    float wb0 = Wih[wsel * 2 + 0];
    float wb1 = Wih[wsel * 2 + 1];
    v16h bmat = {};
    bmat[0] = (_Float16)((lane < 8) ? wb0 : 0.0f);
    bmat[1] = (_Float16)((lane < 8) ? wb1 : 0.0f);

    // Bias folded into the f32 accumulator: c[r] = b[N] for every row M.
    float bias = bih[wsel] + bhh[wsel];
    bias = (col < 8) ? bias : 0.0f;
    v8f cinit;
#pragma unroll
    for (int r = 0; r < 8; ++r) cinit[r] = bias;

    // 256 M-tiles of 16 timesteps, 32 tiles per wave (8 waves).
    for (int tile = wave; tile < T_LEN / 16; tile += 8) {
      const int t0  = tile * 16;
      const int row = t0 + col;
      // A (M x K = 16 x 32): lane m holds row M=m (lanes 0..15), elems K 0..15.
      float a0 = hseq[row * 2 + 0];
      float a1 = hseq[row * 2 + 1];
      const bool lowHalf = (lane < 16);
      v16h amat = {};
      amat[0] = (_Float16)(lowHalf ? a0 : 0.0f);
      amat[1] = (_Float16)(lowHalf ? a1 : 0.0f);

      v8f d = __builtin_amdgcn_wmma_f32_16x16x32_f16(
          false, amat, false, bmat, (short)0, cinit, false, false);

      // D layout: VGPR r on lane L -> (M = r + (L<16 ? 0 : 8), N = L%16)
      if (col < 8) {
        const int base = lowHalf ? t0 : (t0 + 8);
#pragma unroll
        for (int r = 0; r < 8; ++r) xp[(base + r) * 8 + col] = d[r];
      }
    }
    __syncthreads();
    if (tid == 0) lstm_recurrence(xp, hseq, Whh_r + l * 16, &h0, &h1);
    __syncthreads();
  }

  // ---- softmax over the final hidden state (2 classes)
  if (tid == 0) {
    float m  = fmaxf(h0, h1);
    float e0 = fast_exp(h0 - m);
    float e1 = fast_exp(h1 - m);
    float r  = __builtin_amdgcn_rcpf(e0 + e1);
    out[0] = e0 * r;
    out[1] = e1 * r;
  }
}

extern "C" void kernel_launch(void* const* d_in, const int* in_sizes, int n_in,
                              void* d_out, int out_size, void* d_ws, size_t ws_size,
                              hipStream_t stream) {
  (void)in_sizes; (void)n_in; (void)out_size; (void)d_ws; (void)ws_size;
  const float* x     = (const float*)d_in[0];
  // d_in[1] (hidden) is ignored by the reference forward.
  const float* Wih0  = (const float*)d_in[2];
  const float* Whh0  = (const float*)d_in[3];
  const float* bih0  = (const float*)d_in[4];
  const float* bhh0  = (const float*)d_in[5];
  const float* Wih_r = (const float*)d_in[6];
  const float* Whh_r = (const float*)d_in[7];
  const float* bih_r = (const float*)d_in[8];
  const float* bhh_r = (const float*)d_in[9];
  float* out = (float*)d_out;

  const size_t lds_bytes = (size_t)(T_LEN * 8 + T_LEN * 2) * sizeof(float);  // 160 KiB of 320 KiB WGP LDS
  lstm5_kernel<<<dim3(1), dim3(256), lds_bytes, stream>>>(
      x, Wih0, Whh0, bih0, bhh0, Wih_r, Whh_r, bih_r, bhh_r, out);
}
